// Fingerprint_1056561955375
// MI455X (gfx1250) — matrix operand order repaired and checked by reading, not compiled, not run on hardware
//
#include <hip/hip_runtime.h>
#include <hip/hip_bf16.h>

typedef _Float16 v16h __attribute__((ext_vector_type(16)));
typedef _Float16 v8h  __attribute__((ext_vector_type(8)));
typedef float    v8f  __attribute__((ext_vector_type(8)));
typedef float    v4f  __attribute__((ext_vector_type(4)));

namespace {

constexpr int cB = 256, cL = 128, cA = 39, cBD = 10, cK = 6, cM = 256, cFP = 256;
constexpr int BL   = cB * cL;            // 32768 atom rows
constexpr int BLF  = BL * cFP;           // 8388608
constexpr int BLK  = BL * cK;            // 196608 neighbor rows
constexpr long NFE = (long)BLK * cFP;    // 50331648 elements (201MB)
constexpr int G3   = 3 * cFP;            // 768
constexpr int KP0  = 64;                 // padded K for atom_fc / neighbor_fc

__device__ __forceinline__ float wave_sum(float v) {
  #pragma unroll
  for (int off = 16; off > 0; off >>= 1) v += __shfl_xor(v, off, 32);
  return v;
}
__device__ __forceinline__ float sigmoidf_(float x) { return 1.0f / (1.0f + expf(-x)); }

// ---------------------------------------------------------------------------
// Weight convert + zero-pad to f16 [N][Kpad]
// ---------------------------------------------------------------------------
__global__ void k_conv_pad(const float* __restrict__ src, _Float16* __restrict__ dst,
                           int N, int K, int Kpad) {
  long idx = (long)blockIdx.x * blockDim.x + threadIdx.x;
  long total = (long)N * Kpad;
  if (idx >= total) return;
  int n = (int)(idx / Kpad), k = (int)(idx - (long)n * Kpad);
  dst[idx] = (_Float16)((k < K) ? src[(long)n * K + k] : 0.0f);
}

// Pad atom_list rows 39 -> 64 (zero fill), f32
__global__ void k_pad_atoms(const float* __restrict__ atom_list, float* __restrict__ apad) {
  long idx = (long)blockIdx.x * 256 + threadIdx.x;   // grid = BL*64/256 exact
  int m = (int)(idx >> 6), c = (int)(idx & 63);
  apad[idx] = (c < cA) ? atom_list[(long)m * cA + c] : 0.0f;
}

// ---------------------------------------------------------------------------
// Generic WMMA GEMM: out[m,n] = sum_k A[m,k] * W[n,k] + bias[n]*(rowscale[m])
// A: f32 [M][Kpad] (pre-padded), W: f16 [N][Kpad] (pre-padded).
// Requires M%64==0, N%128==0, Kpad%32==0, 32B-aligned rows. 256 thr / 8 waves.
// DUAL: out=raw, out2=relu(raw).  Else out = RELU ? relu : raw.
// ---------------------------------------------------------------------------
constexpr int BM = 64, BN = 128, BK = 32, SA = 40, SB = 40;

template <bool RELU, bool ROWSCALE, bool DUAL>
__global__ __launch_bounds__(256) void k_gemm_wmma(
    const float* __restrict__ A, int Kpad,
    const _Float16* __restrict__ W,
    const float* __restrict__ bias, const float* __restrict__ rowscale,
    float* __restrict__ out, float* __restrict__ out2, int ldo) {
  __shared__ __align__(16) _Float16 As[BM * SA];
  __shared__ __align__(16) _Float16 Bs[BN * SB];

  const int tid  = threadIdx.x;
  const int lane = tid & 31, wid = tid >> 5;
  const int mi = wid & 3, ni = wid >> 2;          // 4 x 2 wave grid
  const int m0 = blockIdx.y * BM, n0 = blockIdx.x * BN;

  v8f acc[4] = {};

  const int ar = tid >> 2;            // 0..63  (A stage row)
  const int ak = (tid & 3) * 8;       // 0..24  (A stage k)
  const int bn = tid >> 1;            // 0..127 (B stage row)
  const int bk = (tid & 1) * 16;      // 0/16   (B stage k)
  const int hi = lane >> 4;           // lane half
  const int l15 = lane & 15;

  const float*    ap = A + (long)(m0 + ar) * Kpad + ak;   // 32B aligned
  const _Float16* wp = W + (long)(n0 + bn) * Kpad + bk;   // 32B aligned

  for (int k0 = 0; k0 < Kpad; k0 += BK) {
    // ---- stage A: two float4 loads -> 8 halfs -> one b128 LDS store ----
    {
      v4f a0 = *(const v4f*)(ap + k0);
      v4f a1 = *(const v4f*)(ap + k0 + 4);
      v8h ph;
      #pragma unroll
      for (int i = 0; i < 4; ++i) { ph[i] = (_Float16)a0[i]; ph[i + 4] = (_Float16)a1[i]; }
      *(v8h*)&As[ar * SA + ak] = ph;
    }
    // ---- stage B: 16 halfs (two b128) ----
    {
      *(v8h*)&Bs[bn * SB + bk]     = *(const v8h*)(wp + k0);
      *(v8h*)&Bs[bn * SB + bk + 8] = *(const v8h*)(wp + k0 + 8);
    }
    __syncthreads();

    // ---- A fragment (ISA 16-bit A 16x32 layout) ----
    const int mr = mi * 16 + l15;
    v8h alo = *(const v8h*)&As[mr * SA + hi * 8];        // K = 8*hi + e
    v8h ahi = *(const v8h*)&As[mr * SA + 16 + hi * 8];   // K = 16 + 8*hi + e
    v16h af;
    #pragma unroll
    for (int i = 0; i < 8; ++i) { af[i] = alo[i]; af[i + 8] = ahi[i]; }

    #pragma unroll
    for (int j = 0; j < 4; ++j) {
      // ---- B fragment: N = lane&15, K = 8*hi + e (lo), 16 + 8*hi + e (hi) ----
      const int nr = ni * 64 + j * 16 + l15;
      v8h blo = *(const v8h*)&Bs[nr * SB + hi * 8];
      v8h bhi = *(const v8h*)&Bs[nr * SB + 16 + hi * 8];
      v16h bf;
      #pragma unroll
      for (int i = 0; i < 8; ++i) { bf[i] = blo[i]; bf[i + 8] = bhi[i]; }
      acc[j] = __builtin_amdgcn_wmma_f32_16x16x32_f16(
          false, af, false, bf, (short)0, acc[j], false, false);
    }
    __syncthreads();
  }

  // ---- epilogue ----
  #pragma unroll
  for (int j = 0; j < 4; ++j) {
    #pragma unroll
    for (int i = 0; i < 8; ++i) {
      int m = m0 + mi * 16 + i + 8 * hi;
      int n = n0 + ni * 64 + j * 16 + l15;
      float v = acc[j][i];
      if (bias) {
        float b = bias[n];
        if (ROWSCALE) b *= rowscale[m];
        v += b;
      }
      long o = (long)m * ldo + n;
      if (DUAL) { out[o] = v; out2[o] = fmaxf(v, 0.0f); }
      else      { out[o] = RELU ? fmaxf(v, 0.0f) : v; }
    }
  }
}

// ---------------------------------------------------------------------------
// Build concatenated neighbor-fc input at stride 64 (zero padded):
// xcat[row][0:39]=atom nbr, [39:49]=bond nbr, [49:64]=0 ; row=(b*L+l)*K+k
// ---------------------------------------------------------------------------
__global__ void k_build_xcat(const float* __restrict__ atom_list,
                             const float* __restrict__ bond_list,
                             const int* __restrict__ adl, const int* __restrict__ bdl,
                             float* __restrict__ xcat) {
  long idx = (long)blockIdx.x * 256 + threadIdx.x;   // grid = BLK*64/256 exact
  int row = (int)(idx >> 6);
  int c   = (int)(idx & 63);
  int b   = row / (cL * cK);
  float v = 0.0f;
  if (c < cA) {
    int ad = adl[row];
    v = atom_list[((long)b * cL + ad) * cA + c];
  } else if (c < cA + cBD) {
    int bd = bdl[row];
    v = bond_list[((long)b * cM + bd) * cBD + (c - cA)];
  }
  xcat[idx] = v;
}

// neigh[(b,l,k),c4..c4+3] = act[b, adl[b,l,k], c4..c4+3]   (float4)
__global__ void k_gather_act(const float* __restrict__ act, const int* __restrict__ adl,
                             float* __restrict__ neigh) {
  long idx = (long)blockIdx.x * 256 + threadIdx.x;   // grid = NFE/1024 exact
  int row = (int)(idx >> 6);
  int c4  = (int)(idx & 63) << 2;
  int b = row / (cL * cK);
  int ad = adl[row];
  *(v4f*)&neigh[(idx << 2)] = *(const v4f*)&act[(((long)b * cL + ad) << 8) + c4];
}

// ---------------------------------------------------------------------------
// Per-(b,l) align score + softmax over K=6. One wave per (b,l).
// ---------------------------------------------------------------------------
__global__ __launch_bounds__(256) void k_atom_score(
    const float* __restrict__ act, const float* __restrict__ neigh,
    const float* __restrict__ aW /*512*/, const float* __restrict__ ab /*1*/,
    const int* __restrict__ adl,
    float* __restrict__ aw_out, float* __restrict__ sumaw) {
  const int wid = threadIdx.x >> 5, lane = threadIdx.x & 31;
  const int bl = blockIdx.x * 8 + wid;               // grid = BL/8, exact
  const float* arow = act + (long)bl * cFP;
  float sa = 0.0f;
  #pragma unroll
  for (int j = 0; j < 8; ++j) sa += arow[lane + 32 * j] * aW[lane + 32 * j];
  sa = wave_sum(sa);
  const float abv = ab[0];

  float s[cK], am[cK];
  #pragma unroll
  for (int k = 0; k < cK; ++k) {
    const float* nr = neigh + ((long)bl * cK + k) * cFP;
    float t = 0.0f;
    #pragma unroll
    for (int j = 0; j < 8; ++j) t += nr[lane + 32 * j] * aW[cFP + lane + 32 * j];
    t = wave_sum(t);
    int ad = adl[bl * cK + k];
    float pad = (ad == cL - 1) ? 1.0f : 0.0f;
    am[k] = 1.0f - pad;
    s[k] = fmaxf(sa + t + abv, 0.0f) - 9.0f * pad;
  }
  float mx = s[0];
  #pragma unroll
  for (int k = 1; k < cK; ++k) mx = fmaxf(mx, s[k]);
  float e[cK], den = 0.0f;
  #pragma unroll
  for (int k = 0; k < cK; ++k) { e[k] = expf(s[k] - mx); den += e[k]; }
  const float inv = 1.0f / den;
  float sm = 0.0f;
  #pragma unroll
  for (int k = 0; k < cK; ++k) { e[k] = e[k] * inv * am[k]; sm += e[k]; }
  if (lane < cK) aw_out[(long)bl * cK + lane] = e[lane];
  if (lane == 0) sumaw[bl] = sm;
}

// wn[b,l,c4..] = sum_k aw[b,l,k] * neigh[b,l,k,c4..]   (float4)
__global__ void k_weighted_neigh(const float* __restrict__ neigh,
                                 const float* __restrict__ aw,
                                 float* __restrict__ wn) {
  long idx = (long)blockIdx.x * 256 + threadIdx.x;   // grid = BLF/1024 exact
  int bl = (int)(idx >> 6);
  int c4 = (int)(idx & 63) << 2;
  v4f acc = {};
  #pragma unroll
  for (int k = 0; k < cK; ++k) {
    float w = aw[(long)bl * cK + k];
    v4f nv = *(const v4f*)&neigh[(((long)bl * cK + k) << 8) + c4];
    acc += w * nv;
  }
  *(v4f*)&wn[idx << 2] = acc;
}

// GRU combine (float4): h' = (1-z)*n + z*h ; act = relu(h')
__global__ void k_gru_combine(const float* __restrict__ gi, const float* __restrict__ gh,
                              const float* __restrict__ hprev,
                              float* __restrict__ hout, float* __restrict__ actout) {
  long idx = (long)blockIdx.x * 256 + threadIdx.x;   // grid = BLF/1024 exact
  long m = idx >> 6;
  int  f = (int)(idx & 63) << 2;
  long base = m * G3 + f;
  v4f ir = *(const v4f*)&gi[base];
  v4f iz = *(const v4f*)&gi[base + cFP];
  v4f in_ = *(const v4f*)&gi[base + 2 * cFP];
  v4f hr = *(const v4f*)&gh[base];
  v4f hz = *(const v4f*)&gh[base + cFP];
  v4f hn = *(const v4f*)&gh[base + 2 * cFP];
  v4f hp = *(const v4f*)&hprev[idx << 2];
  v4f hnew, anew;
  #pragma unroll
  for (int i = 0; i < 4; ++i) {
    float r = sigmoidf_(ir[i] + hr[i]);
    float z = sigmoidf_(iz[i] + hz[i]);
    float n = tanhf(in_[i] + r * hn[i]);
    float h2 = (1.0f - z) * n + z * hp[i];
    hnew[i] = h2;
    anew[i] = fmaxf(h2, 0.0f);
  }
  *(v4f*)&hout[idx << 2]   = hnew;
  *(v4f*)&actout[idx << 2] = anew;
}

// Initial molecule sums over L
__global__ void k_mol_init(const float* __restrict__ h, const float* __restrict__ act,
                           const float* __restrict__ mask,
                           float* __restrict__ molunb0, float* __restrict__ mfv0,
                           float* __restrict__ molfeat, float* __restrict__ actmol) {
  int idx = blockIdx.x * 256 + threadIdx.x;          // grid = B*FP/256 exact
  int b = idx >> 8, f = idx & 255;
  float s1 = 0.0f, s2 = 0.0f;
  for (int l = 0; l < cL; ++l) {
    float m = mask[b * cL + l];
    long o = ((long)b * cL + l) * cFP + f;
    s1 += h[o] * m;
    s2 += act[o] * m;
  }
  molunb0[idx] = s1;
  mfv0[idx] = s2;
  molfeat[idx] = s2;
  actmol[idx] = fmaxf(s2, 0.0f);
}

// Molecule attention: one block per batch b.
__global__ __launch_bounds__(256) void k_mol_score(
    const float* __restrict__ act, const float* __restrict__ actmol,
    const float* __restrict__ mW /*512*/, const float* __restrict__ mb /*1*/,
    const float* __restrict__ mask,
    float* __restrict__ mw_out, float* __restrict__ wact, float* __restrict__ summw) {
  __shared__ float sn[cL];
  __shared__ float mwv[cL];
  __shared__ float sam;
  const int b = blockIdx.x;
  const int tid = threadIdx.x, wid = tid >> 5, lane = tid & 31;

  for (int l = wid; l < cL; l += 8) {
    const float* ar = act + ((long)b * cL + l) * cFP;
    float t = 0.0f;
    #pragma unroll
    for (int j = 0; j < 8; ++j) t += ar[lane + 32 * j] * mW[cFP + lane + 32 * j];
    t = wave_sum(t);
    if (lane == 0) sn[l] = t;
  }
  if (wid == 0) {
    const float* am = actmol + (long)b * cFP;
    float t = 0.0f;
    #pragma unroll
    for (int j = 0; j < 8; ++j) t += am[lane + 32 * j] * mW[lane + 32 * j];
    t = wave_sum(t);
    if (lane == 0) sam = t;
  }
  __syncthreads();
  if (tid == 0) {
    float mbv = mb[0], mx = -1e30f;
    for (int l = 0; l < cL; ++l) {
      float msk = mask[b * cL + l];
      float v = fmaxf(sam + sn[l] + mbv, 0.0f) + ((msk == 0.0f) ? -9e8f : 0.0f);
      sn[l] = v;
      mx = fmaxf(mx, v);
    }
    float den = 0.0f;
    for (int l = 0; l < cL; ++l) den += expf(sn[l] - mx);
    float smw = 0.0f, inv = 1.0f / den;
    for (int l = 0; l < cL; ++l) {
      float w = expf(sn[l] - mx) * inv * mask[b * cL + l];
      mwv[l] = w;
      mw_out[(long)b * cL + l] = w;
      smw += w;
    }
    summw[b] = smw;
  }
  __syncthreads();
  const int c = tid;
  float acc = 0.0f;
  for (int l = 0; l < cL; ++l) acc += mwv[l] * act[((long)b * cL + l) * cFP + c];
  wact[(long)b * cFP + c] = acc;
}

__global__ void k_mol_combine(const float* __restrict__ gi, const float* __restrict__ gh,
                              const float* __restrict__ hprev,
                              float* __restrict__ molfeat, float* __restrict__ actmol,
                              float* __restrict__ molunb_slot, float* __restrict__ mfv_slot,
                              float* __restrict__ final_out) {
  int idx = blockIdx.x * 256 + threadIdx.x;          // B*FP exact
  int m = idx >> 8, f = idx & 255;
  long base = (long)m * G3 + f;
  float ir = gi[base], iz = gi[base + cFP], in_ = gi[base + 2 * cFP];
  float hr = gh[base], hz = gh[base + cFP], hn = gh[base + 2 * cFP];
  float hp = hprev[idx];
  float r = sigmoidf_(ir + hr);
  float z = sigmoidf_(iz + hz);
  float n = tanhf(in_ + r * hn);
  float hnew = (1.0f - z) * n + z * hp;
  float a = fmaxf(hnew, 0.0f);
  molfeat[idx] = hnew;
  molunb_slot[idx] = hnew;
  actmol[idx] = a;
  mfv_slot[idx] = a;
  if (final_out) final_out[idx] = hnew;
}

inline int cdiv(long a, long b) { return (int)((a + b - 1) / b); }

}  // namespace

// ---------------------------------------------------------------------------
extern "C" void kernel_launch(void* const* d_in, const int* in_sizes, int n_in,
                              void* d_out, int out_size, void* d_ws, size_t ws_size,
                              hipStream_t stream) {
  (void)in_sizes; (void)n_in; (void)out_size; (void)ws_size;

  const float* atom_list   = (const float*)d_in[0];
  const float* bond_list   = (const float*)d_in[1];
  const int*   adl         = (const int*)d_in[2];
  const int*   bdl         = (const int*)d_in[3];
  const float* atom_mask   = (const float*)d_in[4];
  const float* atom_fc_W   = (const float*)d_in[5];
  const float* atom_fc_b   = (const float*)d_in[6];
  const float* nei_fc_W    = (const float*)d_in[7];
  const float* nei_fc_b    = (const float*)d_in[8];
  const float* gru_w_ih    = (const float*)d_in[9];
  const float* gru_w_hh    = (const float*)d_in[10];
  const float* gru_b_ih    = (const float*)d_in[11];
  const float* gru_b_hh    = (const float*)d_in[12];
  const float* align_W     = (const float*)d_in[13];
  const float* align_b     = (const float*)d_in[14];
  const float* attend_W    = (const float*)d_in[15];
  const float* attend_b    = (const float*)d_in[16];
  const float* mgru_w_ih   = (const float*)d_in[17];
  const float* mgru_w_hh   = (const float*)d_in[18];
  const float* mgru_b_ih   = (const float*)d_in[19];
  const float* mgru_b_hh   = (const float*)d_in[20];
  const float* mol_align_W = (const float*)d_in[21];
  const float* mol_align_b = (const float*)d_in[22];
  const float* mol_att_W   = (const float*)d_in[23];
  const float* mol_att_b   = (const float*)d_in[24];

  float* out = (float*)d_out;
  // output layout (floats)
  const long o_atomfeat = 0;
  const long o_afv      = (long)BLF;                    // 4 slots
  const long o_awv      = o_afv + 4L * BLF;             // 3 slots of BLK
  const long o_mfv      = o_awv + 3L * BLK;             // 3 slots of B*FP
  const long o_molunb   = o_mfv + 3L * cB * cFP;        // 3 slots of B*FP
  const long o_mawv     = o_molunb + 3L * cB * cFP;     // 2 slots of B*L
  const long o_molfinal = o_mawv + 2L * BL;

  // workspace carve-out
  char* base = (char*)d_ws;
  size_t off = 0;
  auto carve = [&](size_t bytes) { void* p = base + off; off += (bytes + 255) & ~(size_t)255; return p; };

  float* ws_nf    = (float*)carve((size_t)NFE * 4);     // neigh; later gi||gh (exact fit)
  float* gi       = ws_nf;
  float* gh       = ws_nf + (long)BL * G3;
  float* ws_act0  = (float*)carve((size_t)BLF * 4);
  float* ws_wn    = (float*)carve((size_t)BLF * 4);
  float* ws_ctx   = (float*)carve((size_t)BLF * 4);
  float* ws_apad  = (float*)carve((size_t)BL * KP0 * 4);
  float* ws_xcat  = (float*)carve((size_t)BLK * KP0 * 4);
  float* ws_sumaw = (float*)carve((size_t)BL * 4);
  float* ws_mfeat = (float*)carve((size_t)cB * cFP * 4);
  float* ws_amol  = (float*)carve((size_t)cB * cFP * 4);
  float* ws_wact  = (float*)carve((size_t)cB * cFP * 4);
  float* ws_summw = (float*)carve((size_t)cB * 4);
  float* ws_mgi   = (float*)carve((size_t)cB * G3 * 4);
  float* ws_mgh   = (float*)carve((size_t)cB * G3 * 4);
  float* ws_mctx  = (float*)carve((size_t)cB * cFP * 4);

  _Float16* Wat_h   = (_Float16*)carve((size_t)cFP * KP0 * 2);
  _Float16* Wne_h   = (_Float16*)carve((size_t)cFP * KP0 * 2);
  _Float16* Watt_h  = (_Float16*)carve((size_t)3 * cFP * cFP * 2);
  _Float16* Wih_h   = (_Float16*)carve((size_t)3 * G3 * cFP * 2);
  _Float16* Whh_h   = (_Float16*)carve((size_t)3 * G3 * cFP * 2);
  _Float16* Wmatt_h = (_Float16*)carve((size_t)cFP * cFP * 2);
  _Float16* Wmih_h  = (_Float16*)carve((size_t)G3 * cFP * 2);
  _Float16* Wmhh_h  = (_Float16*)carve((size_t)G3 * cFP * 2);

  // ---- weight conversion / padding ----
  auto conv = [&](const float* s, _Float16* d, int N, int K, int Kp) {
    k_conv_pad<<<cdiv((long)N * Kp, 256), 256, 0, stream>>>(s, d, N, K, Kp);
  };
  conv(atom_fc_W, Wat_h, cFP, cA, KP0);
  conv(nei_fc_W, Wne_h, cFP, cA + cBD, KP0);
  conv(attend_W, Watt_h, 3 * cFP, cFP, cFP);
  conv(gru_w_ih, Wih_h, 3 * G3, cFP, cFP);
  conv(gru_w_hh, Whh_h, 3 * G3, cFP, cFP);
  conv(mol_att_W, Wmatt_h, cFP, cFP, cFP);
  conv(mgru_w_ih, Wmih_h, G3, cFP, cFP);
  conv(mgru_w_hh, Wmhh_h, G3, cFP, cFP);

  k_pad_atoms<<<BL * KP0 / 256, 256, 0, stream>>>(atom_list, ws_apad);

  // ---- atom_fc: afv[0] = pre-relu, ws_act0 = relu (initial h == initial act) ----
  k_gemm_wmma<false, false, true><<<dim3(cFP / BN, BL / BM), 256, 0, stream>>>(
      ws_apad, KP0, Wat_h, atom_fc_b, nullptr, out + o_afv, ws_act0, cFP);

  // ---- neighbor_fc: gather concat (stride 64) then GEMM+relu -> nf ----
  k_build_xcat<<<(int)((long)BLK * KP0 / 256), 256, 0, stream>>>(
      atom_list, bond_list, adl, bdl, ws_xcat);
  k_gemm_wmma<true, false, false><<<dim3(cFP / BN, BLK / BM), 256, 0, stream>>>(
      ws_xcat, KP0, Wne_h, nei_fc_b, nullptr, ws_nf, nullptr, cFP);

  // ---- R rounds of neighborhood attention + GRU ----
  const float* cur_act = ws_act0;
  const float* cur_h   = ws_act0;
  for (int d = 0; d < 3; ++d) {
    if (d > 0)
      k_gather_act<<<(int)(NFE / 1024), 256, 0, stream>>>(cur_act, adl, ws_nf);

    k_atom_score<<<BL / 8, 256, 0, stream>>>(
        cur_act, ws_nf, align_W + (long)d * 2 * cFP, align_b + d, adl,
        out + o_awv + (long)d * BLK, ws_sumaw);

    k_weighted_neigh<<<BLF / 1024, 256, 0, stream>>>(
        ws_nf, out + o_awv + (long)d * BLK, ws_wn);

    // ctx = relu(wn @ attend_W^T + sum(aw)*attend_b)
    k_gemm_wmma<true, true, false><<<dim3(cFP / BN, BL / BM), 256, 0, stream>>>(
        ws_wn, cFP, Watt_h + (long)d * cFP * cFP,
        attend_b + (long)d * cFP, ws_sumaw, ws_ctx, nullptr, cFP);

    // gi = ctx @ w_ih^T + b_ih ; gh = h @ w_hh^T + b_hh   (overlaid on nf buffer)
    k_gemm_wmma<false, false, false><<<dim3(G3 / BN, BL / BM), 256, 0, stream>>>(
        ws_ctx, cFP, Wih_h + (long)d * G3 * cFP,
        gru_b_ih + (long)d * G3, nullptr, gi, nullptr, G3);
    k_gemm_wmma<false, false, false><<<dim3(G3 / BN, BL / BM), 256, 0, stream>>>(
        cur_h, cFP, Whh_h + (long)d * G3 * cFP,
        gru_b_hh + (long)d * G3, nullptr, gh, nullptr, G3);

    float* act_out = out + o_afv + (long)(d + 1) * BLF;
    k_gru_combine<<<BLF / 1024, 256, 0, stream>>>(
        gi, gh, cur_h, out + o_atomfeat, act_out);

    cur_h   = out + o_atomfeat;
    cur_act = act_out;
  }

  // ---- molecule phase ----
  k_mol_init<<<cB * cFP / 256, 256, 0, stream>>>(
      out + o_atomfeat, cur_act, atom_mask,
      out + o_molunb, out + o_mfv, ws_mfeat, ws_amol);

  for (int t = 0; t < 2; ++t) {
    k_mol_score<<<cB, 256, 0, stream>>>(
        cur_act, ws_amol, mol_align_W, mol_align_b, atom_mask,
        out + o_mawv + (long)t * BL, ws_wact, ws_summw);

    // mctx = relu(wact @ mol_attend_W^T + sum(mw)*mol_attend_b)
    k_gemm_wmma<true, true, false><<<dim3(cFP / BN, cB / BM), 256, 0, stream>>>(
        ws_wact, cFP, Wmatt_h, mol_att_b, ws_summw, ws_mctx, nullptr, cFP);

    k_gemm_wmma<false, false, false><<<dim3(G3 / BN, cB / BM), 256, 0, stream>>>(
        ws_mctx, cFP, Wmih_h, mgru_b_ih, nullptr, ws_mgi, nullptr, G3);
    k_gemm_wmma<false, false, false><<<dim3(G3 / BN, cB / BM), 256, 0, stream>>>(
        ws_mfeat, cFP, Wmhh_h, mgru_b_hh, nullptr, ws_mgh, nullptr, G3);

    k_mol_combine<<<cB * cFP / 256, 256, 0, stream>>>(
        ws_mgi, ws_mgh, ws_mfeat, ws_mfeat, ws_amol,
        out + o_molunb + (long)(t + 1) * cB * cFP,
        out + o_mfv + (long)(t + 1) * cB * cFP,
        (t == 1) ? (out + o_molfinal) : nullptr);
  }
}
